// MadeBernoulli_59622736003785
// MI455X (gfx1250) — compile-verified, matrix operand order checked by reading, not run on hardware
//
#include <hip/hip_runtime.h>
#include <hip/hip_bf16.h>
#include <stdint.h>

typedef __attribute__((ext_vector_type(16))) __bf16 v16bf;
typedef __attribute__((ext_vector_type(8)))  __bf16 v8bf;
typedef __attribute__((ext_vector_type(8)))  float  v8f;

#define D_DIM 2048
#define H_DIM 4096
#define B_DIM 8192
#define DEG_MOD 2047   // hid_deg[h] = h % 2047 + 1

constexpr int BM = 128, BN = 128, BK = 32, LDSK = 40; // LDSK padded to spread LDS banks

// CDNA5 async global->LDS copy (ASYNCcnt-tracked), per ISA 15.18.3 / CDNA5 ISA.
// LDS operand is a per-lane VGPR holding the LDS byte address (generic-pointer low 32 bits).
#define USE_ASYNC_G2L 1

#if USE_ASYNC_G2L
__device__ __forceinline__ void async_g2l_b128(const void* g, void* l) {
  asm volatile("global_load_async_to_lds_b128 %0, %1, off"
               :: "v"((uint32_t)(uintptr_t)l), "v"((uint64_t)(uintptr_t)g)
               : "memory");
}
__device__ __forceinline__ void wait_async0() {
  asm volatile("s_wait_asynccnt 0x0" ::: "memory");
}
#endif

// ---------------- small prep kernels ----------------

__global__ __launch_bounds__(256) void zero_out_k(float* out, int n) {
  int i = blockIdx.x * blockDim.x + threadIdx.x;
  if (i < n) out[i] = 0.f;
}

__global__ __launch_bounds__(256) void cvt_value_k(const float* __restrict__ v,
                                                   __bf16* __restrict__ o, int n) {
  int i = (blockIdx.x * blockDim.x + threadIdx.x) * 4;
  if (i < n) {
    float4 f = *(const float4*)(v + i);
    o[i + 0] = (__bf16)f.x; o[i + 1] = (__bf16)f.y;
    o[i + 2] = (__bf16)f.z; o[i + 3] = (__bf16)f.w;
  }
}

// W1: (H,D), mask = (h%2047 >= d)
__global__ __launch_bounds__(256) void mask_w1_k(const float* __restrict__ W,
                                                 __bf16* __restrict__ o) {
  int d = blockIdx.x * 256 + threadIdx.x;
  int h = blockIdx.y;
  size_t idx = (size_t)h * D_DIM + d;
  o[idx] = ((h % DEG_MOD) >= d) ? (__bf16)W[idx] : (__bf16)0.f;
}

// W2: (H,H), mask = (h%2047 >= c%2047)
__global__ __launch_bounds__(256) void mask_w2_k(const float* __restrict__ W,
                                                 __bf16* __restrict__ o) {
  int c = blockIdx.x * 256 + threadIdx.x;
  int h = blockIdx.y;
  size_t idx = (size_t)h * H_DIM + c;
  o[idx] = ((h % DEG_MOD) >= (c % DEG_MOD)) ? (__bf16)W[idx] : (__bf16)0.f;
}

// W3: (D,H), mask = (d > h%2047)
__global__ __launch_bounds__(256) void mask_w3_k(const float* __restrict__ W,
                                                 __bf16* __restrict__ o) {
  int c = blockIdx.x * 256 + threadIdx.x;
  int d = blockIdx.y;
  size_t idx = (size_t)d * H_DIM + c;
  o[idx] = (d > (c % DEG_MOD)) ? (__bf16)W[idx] : (__bf16)0.f;
}

// ---------------- WMMA GEMM:  out[m][n] = act( sum_k A[m][k]*W[n][k] + bias[n] ) ----------------
// FINAL==0: relu -> coalesced bf16 store via LDS-staged C tile.
// FINAL==1: fused log-prob epilogue -> shuffle-reduce + atomicAdd per row.

template<int FINAL>
__global__ __launch_bounds__(256)
void made_gemm(const __bf16* __restrict__ A, const __bf16* __restrict__ W,
               const float* __restrict__ bias,
               int M, int N, int K,
               __bf16* __restrict__ outBf,
               const float* __restrict__ value,
               float* __restrict__ lp)
{
  __shared__ alignas(16) __bf16 smem[2 * BM * LDSK + 2 * BN * LDSK];
  __bf16* sA = smem;                       // [2][BM][LDSK]
  __bf16* sW = smem + 2 * BM * LDSK;       // [2][BN][LDSK]

  const int tid   = threadIdx.x;
  const int lane  = tid & 31;
  const int wave  = tid >> 5;
  const int waveM = wave >> 2;             // 0..1 -> 64 rows each
  const int waveN = wave & 3;              // 0..3 -> 32 cols each
  const int blockM = blockIdx.y * BM;
  const int blockN = blockIdx.x * BN;

  // cooperative tile load mapping: 256 threads x 2 chunks of 8 bf16 per matrix
  const int r0 = tid >> 2;                 // 0..63
  const int c0 = (tid & 3) * 8;            // 0,8,16,24

  v8f acc[4][2];
  const v8f vzero = {0.f,0.f,0.f,0.f,0.f,0.f,0.f,0.f};
  #pragma unroll
  for (int i = 0; i < 4; ++i)
    #pragma unroll
    for (int j = 0; j < 2; ++j) acc[i][j] = vzero;

#if USE_ASYNC_G2L
  auto load_tiles = [&](int kt, int buf) {
    const int kb = kt * BK;
    async_g2l_b128(A + (size_t)(blockM      + r0) * K + kb + c0, &sA[(buf * BM +      r0) * LDSK + c0]);
    async_g2l_b128(A + (size_t)(blockM + 64 + r0) * K + kb + c0, &sA[(buf * BM + 64 + r0) * LDSK + c0]);
    async_g2l_b128(W + (size_t)(blockN      + r0) * K + kb + c0, &sW[(buf * BN +      r0) * LDSK + c0]);
    async_g2l_b128(W + (size_t)(blockN + 64 + r0) * K + kb + c0, &sW[(buf * BN + 64 + r0) * LDSK + c0]);
  };
#else
  v8bf ra0, ra1, rw0, rw1;
  auto load_global = [&](int kt) {
    const int kb = kt * BK;
    ra0 = *(const v8bf*)(A + (size_t)(blockM      + r0) * K + kb + c0);
    ra1 = *(const v8bf*)(A + (size_t)(blockM + 64 + r0) * K + kb + c0);
    rw0 = *(const v8bf*)(W + (size_t)(blockN      + r0) * K + kb + c0);
    rw1 = *(const v8bf*)(W + (size_t)(blockN + 64 + r0) * K + kb + c0);
  };
  auto store_lds = [&](int buf) {
    *(v8bf*)&sA[(buf * BM +      r0) * LDSK + c0] = ra0;
    *(v8bf*)&sA[(buf * BM + 64 + r0) * LDSK + c0] = ra1;
    *(v8bf*)&sW[(buf * BN +      r0) * LDSK + c0] = rw0;
    *(v8bf*)&sW[(buf * BN + 64 + r0) * LDSK + c0] = rw1;
  };
#endif

  union Frag { v16bf f; v8bf h[2]; };

  auto compute = [&](int buf) {
    const int mr = lane & 15;
    const int kh = (lane >> 4) * 8;        // ISA A/B layout: K 0..7|16..23 vs 8..15|24..31
    v16bf afrag[4], bfrag[2];
    #pragma unroll
    for (int i = 0; i < 4; ++i) {
      const __bf16* p = &sA[(buf * BM + waveM * 64 + i * 16 + mr) * LDSK + kh];
      Frag f; f.h[0] = *(const v8bf*)p; f.h[1] = *(const v8bf*)(p + 16);
      afrag[i] = f.f;
    }
    #pragma unroll
    for (int j = 0; j < 2; ++j) {
      const __bf16* p = &sW[(buf * BN + waveN * 32 + j * 16 + mr) * LDSK + kh];
      Frag f; f.h[0] = *(const v8bf*)p; f.h[1] = *(const v8bf*)(p + 16);
      bfrag[j] = f.f;
    }
    #pragma unroll
    for (int i = 0; i < 4; ++i)
      #pragma unroll
      for (int j = 0; j < 2; ++j)
        acc[i][j] = __builtin_amdgcn_wmma_f32_16x16x32_bf16(
            false, afrag[i], false, bfrag[j], (short)0, acc[i][j], false, false);
  };

  const int KT = K / BK;

#if USE_ASYNC_G2L
  load_tiles(0, 0);
  wait_async0();
  __syncthreads();
  for (int kt = 0; kt < KT; ++kt) {
    const int buf = kt & 1;
    const bool more = (kt + 1 < KT);
    if (more) load_tiles(kt + 1, buf ^ 1);   // async, overlaps WMMA below
    compute(buf);
    if (more) wait_async0();                 // LDS writes visible before barrier
    __syncthreads();
  }
#else
  load_global(0);
  store_lds(0);
  __syncthreads();
  for (int kt = 0; kt < KT; ++kt) {
    const int buf = kt & 1;
    const bool more = (kt + 1 < KT);
    if (more) load_global(kt + 1);
    compute(buf);
    if (more) store_lds(buf ^ 1);
    __syncthreads();
  }
#endif

  // ---------------- epilogue ----------------
  if (FINAL == 0) {
    // Stage C tile in LDS (reuse tile buffer), then fully coalesced b128 stores.
    __bf16* sC = smem;                     // BM*BN bf16 = 32KB <= 40KB tile buffer
    #pragma unroll
    for (int i = 0; i < 4; ++i) {
      const int rowIn = waveM * 64 + i * 16 + ((lane >> 4) * 8);
      #pragma unroll
      for (int j = 0; j < 2; ++j) {
        const int colIn = waveN * 32 + j * 16 + (lane & 15);
        const float bj = bias[blockN + colIn];
        #pragma unroll
        for (int v = 0; v < 8; ++v) {
          float val = acc[i][j][v] + bj;
          val = fmaxf(val, 0.f);
          sC[(rowIn + v) * BN + colIn] = (__bf16)val;
        }
      }
    }
    __syncthreads();
    const int rr = tid >> 1;               // 0..127
    const int cb = (tid & 1) * 64;         // 0 or 64
    #pragma unroll
    for (int s = 0; s < 8; ++s)
      *(v8bf*)(outBf + (size_t)(blockM + rr) * N + blockN + cb + s * 8) =
          *(const v8bf*)&sC[rr * BN + cb + s * 8];
  } else {
    // log_prob contribution: value*t - softplus(t), reduce 16 lanes -> atomicAdd per row
    #pragma unroll
    for (int i = 0; i < 4; ++i) {
      const int rowBase = blockM + waveM * 64 + i * 16 + ((lane >> 4) * 8);
      #pragma unroll
      for (int v = 0; v < 8; ++v) {
        const int row = rowBase + v;
        float lpsum = 0.f;
        #pragma unroll
        for (int j = 0; j < 2; ++j) {
          const int col = blockN + waveN * 32 + j * 16 + (lane & 15);
          float t = acc[i][j][v] + bias[col];
          float sp = fmaxf(t, 0.f) + log1pf(expf(-fabsf(t)));  // stable softplus
          lpsum += value[(size_t)row * D_DIM + col] * t - sp;
        }
        #pragma unroll
        for (int off = 1; off < 16; off <<= 1)
          lpsum += __shfl_xor(lpsum, off, 32);
        if ((lane & 15) == 0) atomicAdd(&lp[row], lpsum);
      }
    }
  }
}

// ---------------- host launcher ----------------

extern "C" void kernel_launch(void* const* d_in, const int* in_sizes, int n_in,
                              void* d_out, int out_size, void* d_ws, size_t ws_size,
                              hipStream_t stream) {
  (void)in_sizes; (void)n_in; (void)out_size; (void)ws_size;
  const float* value = (const float*)d_in[0];
  const float* W1 = (const float*)d_in[1];
  const float* b1 = (const float*)d_in[2];
  const float* W2 = (const float*)d_in[3];
  const float* b2 = (const float*)d_in[4];
  const float* W3 = (const float*)d_in[5];
  const float* b3 = (const float*)d_in[6];
  float* out = (float*)d_out;

  char* ws = (char*)d_ws;
  __bf16* vbf  = (__bf16*)ws;  ws += (size_t)B_DIM * D_DIM * 2;
  __bf16* w1bf = (__bf16*)ws;  ws += (size_t)H_DIM * D_DIM * 2;
  __bf16* w2bf = (__bf16*)ws;  ws += (size_t)H_DIM * H_DIM * 2;
  __bf16* w3bf = (__bf16*)ws;  ws += (size_t)D_DIM * H_DIM * 2;
  __bf16* h1   = (__bf16*)ws;  ws += (size_t)B_DIM * H_DIM * 2;
  __bf16* h2   = (__bf16*)ws;

  zero_out_k<<<(B_DIM + 255) / 256, 256, 0, stream>>>(out, B_DIM);
  cvt_value_k<<<(B_DIM * D_DIM / 4 + 255) / 256, 256, 0, stream>>>(value, vbf, B_DIM * D_DIM);
  mask_w1_k<<<dim3(D_DIM / 256, H_DIM), 256, 0, stream>>>(W1, w1bf);
  mask_w2_k<<<dim3(H_DIM / 256, H_DIM), 256, 0, stream>>>(W2, w2bf);
  mask_w3_k<<<dim3(H_DIM / 256, D_DIM), 256, 0, stream>>>(W3, w3bf);

  // h1 = relu(vbf @ w1bf^T + b1)     : M=8192, N=4096, K=2048
  made_gemm<0><<<dim3(H_DIM / BN, B_DIM / BM), 256, 0, stream>>>(
      vbf, w1bf, b1, B_DIM, H_DIM, D_DIM, h1, nullptr, nullptr);
  // h2 = relu(h1 @ w2bf^T + b2)      : M=8192, N=4096, K=4096
  made_gemm<0><<<dim3(H_DIM / BN, B_DIM / BM), 256, 0, stream>>>(
      h1, w2bf, b2, B_DIM, H_DIM, H_DIM, h2, nullptr, nullptr);
  // logits = h2 @ w3bf^T + b3, fused log-prob reduction : M=8192, N=2048, K=4096
  made_gemm<1><<<dim3(D_DIM / BN, B_DIM / BM), 256, 0, stream>>>(
      h2, w3bf, b3, B_DIM, D_DIM, H_DIM, nullptr, value, out);
}
